// MemoryModule_34033320854152
// MI455X (gfx1250) — matrix operand order repaired and checked by reading, not compiled, not run on hardware
//
#include <hip/hip_runtime.h>
#include <stdint.h>
#include <stddef.h>

// ---------- types ----------
typedef __attribute__((ext_vector_type(16))) __bf16 v16bf;
typedef __attribute__((ext_vector_type(8)))  float  v8f;
typedef __attribute__((ext_vector_type(4)))  int    v4i;
typedef __attribute__((ext_vector_type(4)))  float  v4f;

// ---------- helpers ----------
__device__ __forceinline__ unsigned short f2bf(float f) {
  union { float f; unsigned int u; } c; c.f = f;
  unsigned int x = c.u;
  unsigned int r = (x + 0x7FFFu + ((x >> 16) & 1u)) >> 16;  // round-to-nearest-even
  return (unsigned short)r;
}
__device__ __forceinline__ unsigned int packbf(float lo, float hi) {
  return (unsigned int)f2bf(lo) | ((unsigned int)f2bf(hi) << 16);
}
// Load a 16-element bf16 fragment as two 16-byte chunks separated by `gap` halfs.
__device__ __forceinline__ v16bf frag_gap(const unsigned short* p, int gap) {
  union { v4i i[2]; v16bf b; } u;
  u.i[0] = *reinterpret_cast<const v4i*>(p);
  u.i[1] = *reinterpret_cast<const v4i*>(p + gap);
  return u.b;
}
__device__ __forceinline__ float sigmoidf_(float x) {
  return 1.0f / (1.0f + __expf(-x));
}

#define WMMA_BF16(a, b, c) \
  __builtin_amdgcn_wmma_f32_16x16x32_bf16(false, (a), false, (b), (short)0, (c), false, false)

// dims (fixed by the reference)
#define MEMD 128
#define XD   384          // nf(128) + mem(128) + ef(128)
#define SA   392          // LDS x-tile row stride (halfs), 392*2B = 196 dwords == 4 mod 64 banks
#define SH   136          // LDS h1/msg row stride (halfs)
#define SM   132          // LDS mem-f32 row stride (floats)

// ws layout (halfs)
#define W1OFF  0          // 128*384
#define W2OFF  49152      // 128*128
#define WIHOFF 65536      // 384*128
#define WHHOFF 114688     // 384*128
#define WTOTAL 163840

// ---------- weight f32 -> bf16 conversion ----------
__global__ void cvt_weights_kernel(const float* __restrict__ W1, const float* __restrict__ W2,
                                   const float* __restrict__ Wih, const float* __restrict__ Whh,
                                   unsigned short* __restrict__ ws) {
  int i = blockIdx.x * blockDim.x + threadIdx.x;
  if (i < 49152)        ws[i] = f2bf(W1[i]);
  else if (i < 65536)   ws[i] = f2bf(W2[i - 49152]);
  else if (i < 114688)  ws[i] = f2bf(Wih[i - 65536]);
  else if (i < WTOTAL)  ws[i] = f2bf(Whh[i - 114688]);
}

// ---------- bulk output init: copy memory, copy last_update, zero messages ----------
__global__ void init_out_kernel(const float* __restrict__ memory,
                                const float* __restrict__ last_update,
                                float* __restrict__ out, long nmemv, long nlu) {
  const long i0 = (long)blockIdx.x * blockDim.x + threadIdx.x;
  const long stride = (long)gridDim.x * blockDim.x;
  const long nmem4 = nmemv >> 2, nlu4 = nlu >> 2;
  v4f* out4 = reinterpret_cast<v4f*>(out);
  const v4f* mem4 = reinterpret_cast<const v4f*>(memory);
  for (long i = i0; i < nmem4; i += stride) out4[i] = mem4[i];
  const v4f* lu4 = reinterpret_cast<const v4f*>(last_update);
  v4f* outlu4 = reinterpret_cast<v4f*>(out + nmemv);
  for (long i = i0; i < nlu4; i += stride) outlu4[i] = lu4[i];
  v4f* outmsg4 = reinterpret_cast<v4f*>(out + nmemv + nlu);
  const v4f z = {0.f, 0.f, 0.f, 0.f};
  for (long i = i0; i < nmem4; i += stride) outmsg4[i] = z;
}

// ---------- fused tile kernel: 1 wave32 handles 16 batch rows ----------
__global__ void __launch_bounds__(32)
tgn_tile_kernel(const int*   __restrict__ node_idxs,
                const float* __restrict__ node_features,
                const float* __restrict__ edge_features,
                const float* __restrict__ timestamps,
                const float* __restrict__ memory,
                const float* __restrict__ b1,
                const float* __restrict__ b2,
                const float* __restrict__ b_ih,
                const float* __restrict__ b_hh,
                const unsigned short* __restrict__ wbf,
                float* __restrict__ out_mem,
                float* __restrict__ out_lu,
                float* __restrict__ out_msg) {
  const int lane = threadIdx.x;           // 0..31, wave32
  const int nl = lane & 15;               // column / row-in-tile selector
  const int hf = lane >> 4;               // half-wave selector
  const int rbase = blockIdx.x * 16;

  __shared__ __align__(16) unsigned short s_x[16 * SA];    // bf16 x-tile [16][384]
  __shared__ __align__(16) unsigned short s_h1[16 * SH];   // bf16 h1     [16][128]
  __shared__ __align__(16) unsigned short s_msg[16 * SH];  // bf16 msg    [16][128]
  __shared__ __align__(16) float          s_memf[16 * SM]; // f32 memory  [16][128]
  __shared__ int s_node[16];

  if (lane < 16) s_node[lane] = node_idxs[rbase + lane];

  // ---- stage: gather + concat + bf16 convert into LDS (fully coalesced) ----
#pragma unroll 4
  for (int m = 0; m < 16; ++m) {
    const int gr = rbase + m;
    const int nid = node_idxs[gr];
    const v4f nf = *reinterpret_cast<const v4f*>(node_features + (size_t)gr * MEMD + lane * 4);
    const v4f mv = *reinterpret_cast<const v4f*>(memory        + (size_t)nid * MEMD + lane * 4);
    const v4f ef = *reinterpret_cast<const v4f*>(edge_features + (size_t)gr * MEMD + lane * 4);
    unsigned int* xrow = reinterpret_cast<unsigned int*>(s_x + m * SA);
    xrow[lane * 2 + 0]       = packbf(nf[0], nf[1]);
    xrow[lane * 2 + 1]       = packbf(nf[2], nf[3]);
    xrow[64 + lane * 2 + 0]  = packbf(mv[0], mv[1]);
    xrow[64 + lane * 2 + 1]  = packbf(mv[2], mv[3]);
    xrow[128 + lane * 2 + 0] = packbf(ef[0], ef[1]);
    xrow[128 + lane * 2 + 1] = packbf(ef[2], ef[3]);
    reinterpret_cast<v4f*>(s_memf + m * SM)[lane] = mv;
  }
  // single-wave workgroup: LDS ops are in-order per wave, no barrier needed

  // ---- layer 1: h1 = relu(x @ W1^T + b1), K=384 (12 k-tiles), N=128 (8 n-tiles) ----
#pragma unroll 1
  for (int nt = 0; nt < 8; ++nt) {
    const int n = nt * 16 + nl;
    const float bv = b1[n];
    v8f acc;
#pragma unroll
    for (int r = 0; r < 8; ++r) acc[r] = bv;
    const unsigned short* wp = wbf + W1OFF + (size_t)n * XD + hf * 16;
    const unsigned short* ap = s_x + nl * SA + hf * 16;
#pragma unroll
    for (int kt = 0; kt < 12; ++kt) {
      v16bf a = frag_gap(ap + kt * 32, 8);
      v16bf b = frag_gap(wp + kt * 32, 8);
      acc = WMMA_BF16(a, b, acc);
    }
#pragma unroll
    for (int r = 0; r < 8; ++r) {
      float v = acc[r];
      v = v > 0.f ? v : 0.f;
      s_h1[(r + 8 * hf) * SH + n] = f2bf(v);
    }
  }

  // ---- layer 2: msg = h1 @ W2^T + b2, K=128, N=128; scatter f32 msg to output ----
#pragma unroll 1
  for (int nt = 0; nt < 8; ++nt) {
    const int n = nt * 16 + nl;
    const float bv = b2[n];
    v8f acc;
#pragma unroll
    for (int r = 0; r < 8; ++r) acc[r] = bv;
    const unsigned short* wp = wbf + W2OFF + (size_t)n * MEMD + hf * 16;
    const unsigned short* ap = s_h1 + nl * SH + hf * 16;
#pragma unroll
    for (int kt = 0; kt < 4; ++kt) {
      v16bf a = frag_gap(ap + kt * 32, 8);
      v16bf b = frag_gap(wp + kt * 32, 8);
      acc = WMMA_BF16(a, b, acc);
    }
#pragma unroll
    for (int r = 0; r < 8; ++r) {
      const int m = r + 8 * hf;
      const float v = acc[r];
      s_msg[m * SH + n] = f2bf(v);
      out_msg[(size_t)s_node[m] * MEMD + n] = v;
    }
  }

  // ---- GRU: gi = msg @ W_ih^T + b_ih ; gh = mem @ W_hh^T + b_hh ; gates ----
#pragma unroll 1
  for (int j = 0; j < 8; ++j) {
    const int n = j * 16 + nl;  // gate-column within 128
    const unsigned short* ap_m = s_msg + nl * SH + hf * 16;        // A for gi
    const unsigned short* ap_h = s_x + nl * SA + 128 + hf * 16;    // A for gh (memory cols of x)
    const unsigned short* wih_r = wbf + WIHOFF + (size_t)n * MEMD + hf * 16;
    const unsigned short* whh_r = wbf + WHHOFF + (size_t)n * MEMD + hf * 16;
    const unsigned short* wih_z = wih_r + (size_t)128 * MEMD;
    const unsigned short* whh_z = whh_r + (size_t)128 * MEMD;
    const unsigned short* wih_n = wih_r + (size_t)256 * MEMD;
    const unsigned short* whh_n = whh_r + (size_t)256 * MEMD;

    v8f g_ir, g_hr, g_iz, g_hz, g_in, g_hn;
    {
      const float bir = b_ih[n],       bhr = b_hh[n];
      const float biz = b_ih[128 + n], bhz = b_hh[128 + n];
      const float bin = b_ih[256 + n], bhn = b_hh[256 + n];
#pragma unroll
      for (int r = 0; r < 8; ++r) {
        g_ir[r] = bir; g_hr[r] = bhr; g_iz[r] = biz;
        g_hz[r] = bhz; g_in[r] = bin; g_hn[r] = bhn;
      }
    }
#pragma unroll
    for (int kt = 0; kt < 4; ++kt) {
      v16bf am = frag_gap(ap_m + kt * 32, 8);
      v16bf ah = frag_gap(ap_h + kt * 32, 8);
      g_ir = WMMA_BF16(am, frag_gap(wih_r + kt * 32, 8), g_ir);
      g_hr = WMMA_BF16(ah, frag_gap(whh_r + kt * 32, 8), g_hr);
      g_iz = WMMA_BF16(am, frag_gap(wih_z + kt * 32, 8), g_iz);
      g_hz = WMMA_BF16(ah, frag_gap(whh_z + kt * 32, 8), g_hz);
      g_in = WMMA_BF16(am, frag_gap(wih_n + kt * 32, 8), g_in);
      g_hn = WMMA_BF16(ah, frag_gap(whh_n + kt * 32, 8), g_hn);
    }
#pragma unroll
    for (int r = 0; r < 8; ++r) {
      const int m = r + 8 * hf;
      const float rg = sigmoidf_(g_ir[r] + g_hr[r]);
      const float zg = sigmoidf_(g_iz[r] + g_hz[r]);
      const float ng = tanhf(g_in[r] + rg * g_hn[r]);
      const float mem = s_memf[m * SM + n];
      out_mem[(size_t)s_node[m] * MEMD + n] = (1.0f - zg) * ng + zg * mem;
    }
  }

  // ---- last_update scatter ----
  if (lane < 16) out_lu[s_node[lane]] = timestamps[rbase + lane];
}

// ---------- launch ----------
extern "C" void kernel_launch(void* const* d_in, const int* in_sizes, int n_in,
                              void* d_out, int out_size, void* d_ws, size_t ws_size,
                              hipStream_t stream) {
  const int*   node_idxs     = (const int*)d_in[0];
  const float* node_features = (const float*)d_in[1];
  const float* edge_features = (const float*)d_in[2];
  const float* timestamps    = (const float*)d_in[3];
  const float* memory        = (const float*)d_in[4];
  const float* last_update   = (const float*)d_in[5];
  const float* W1 = (const float*)d_in[6];
  const float* b1 = (const float*)d_in[7];
  const float* W2 = (const float*)d_in[8];
  const float* b2 = (const float*)d_in[9];
  const float* Wih = (const float*)d_in[10];
  const float* bih = (const float*)d_in[11];
  const float* Whh = (const float*)d_in[12];
  const float* bhh = (const float*)d_in[13];

  const int  B = in_sizes[0];          // 100000
  const long N = (long)in_sizes[5];    // 500000
  const long nmemv = N * MEMD;         // 64,000,000

  float* out = (float*)d_out;
  float* out_mem = out;
  float* out_lu  = out + nmemv;
  float* out_msg = out + nmemv + N;
  unsigned short* wbf = (unsigned short*)d_ws;  // 320 KB bf16 weights

  cvt_weights_kernel<<<(WTOTAL + 255) / 256, 256, 0, stream>>>(W1, W2, Wih, Whh, wbf);
  init_out_kernel<<<4096, 256, 0, stream>>>(memory, last_update, out, nmemv, N);
  tgn_tile_kernel<<<B / 16, 32, 0, stream>>>(node_idxs, node_features, edge_features,
                                             timestamps, memory, b1, b2, bih, bhh, wbf,
                                             out_mem, out_lu, out_msg);
}
